// BindingAffinityGNN_42760694399003
// MI455X (gfx1250) — matrix operand order repaired and checked by reading, not compiled, not run on hardware
//
#include <hip/hip_runtime.h>
#include <hip/hip_bf16.h>

#define N_NODES 50000
#define N_EDGES 600000
#define N_GRAPHS 64
#define IN_DIM 6
#define HID 128

typedef __attribute__((ext_vector_type(16))) _Float16 v16h;
typedef __attribute__((ext_vector_type(8)))  _Float16 v8h;
typedef __attribute__((ext_vector_type(8)))  float    v8f;

// ---------------- degree / norm ----------------
__global__ void k_init_deg(float* deg, int n) {
    int i = blockIdx.x * blockDim.x + threadIdx.x;
    if (i < n) deg[i] = 1.0f;  // self-loop contributes 1
}

__global__ void k_deg(const int* __restrict__ ei, float* __restrict__ deg, int E) {
    int e = blockIdx.x * blockDim.x + threadIdx.x;
    if (e < E) atomicAdd(&deg[ei[E + e]], 1.0f);  // dst row of edge_index
}

__global__ void k_dis(const float* __restrict__ deg, float* __restrict__ dis, int n) {
    int i = blockIdx.x * blockDim.x + threadIdx.x;
    if (i < n) dis[i] = rsqrtf(deg[i]);  // deg >= 1 always (self-loop)
}

// ---------------- weight prep: W[k][c] f32 -> Wt[c][k] f16 ----------------
__global__ void k_prep_w(const float* __restrict__ W, _Float16* __restrict__ Wt) {
    int idx = blockIdx.x * blockDim.x + threadIdx.x;  // HID*HID
    if (idx >= HID * HID) return;
    int k = idx >> 7, c = idx & (HID - 1);
    Wt[c * HID + k] = (_Float16)W[k * HID + c];
}

// ---------------- layer 1 GEMM (K=6, scalar) ----------------
__global__ void k_gemm_in(const float* __restrict__ x, const float* __restrict__ W1,
                          float* __restrict__ h, int n) {
    int idx = blockIdx.x * blockDim.x + threadIdx.x;  // n*HID
    if (idx >= n * HID) return;
    int i = idx >> 7, c = idx & (HID - 1);
    float acc = 0.0f;
#pragma unroll
    for (int k = 0; k < IN_DIM; ++k) acc += x[i * IN_DIM + k] * W1[k * HID + c];
    h[idx] = acc;
}

// ---------------- layers 2/3 GEMM: WMMA f16 -> f32, K=128 ----------------
// A: [n][128] f16 row-major.  Bt: [col][k] f16 (transposed W).  C: [n][128] f32.
// One wave per 16x16 tile; 8 waves/block cover all 128 output columns.
__global__ void __launch_bounds__(256)
k_gemm_wmma(const _Float16* __restrict__ A, const _Float16* __restrict__ Bt,
            float* __restrict__ C, int nrows) {
    const int lane = threadIdx.x & 31;
    const int wave = threadIdx.x >> 5;        // tile_n = wave (0..7)
    const int m0   = blockIdx.x * 16;
    if (m0 >= nrows) return;                  // nrows is a multiple of 16 (50000 = 3125*16)
    const int kh   = lane >> 4;               // 0: K-low half of lanes, 1: K-high half
    const int mrow = m0 + (lane & 15);
    const int col  = wave * 16 + (lane & 15);

    const _Float16* arow = A  + (size_t)mrow * HID;
    const _Float16* bcol = Bt + (size_t)col  * HID;

    v8f acc = {};
#pragma unroll
    for (int kb = 0; kb < 4; ++kb) {          // 4 x (16x16x32)
        const int kbase = kb * 32;
        // A k order used here (contiguous, not the ISA table order): lane holds K = kbase+16*kh..+15 (matches B)
        v8h alo = *(const v8h*)(arow + kbase + 16 * kh);
        v8h ahi = *(const v8h*)(arow + kbase + 16 * kh + 8);
        v16h afrag;
#pragma unroll
        for (int i = 0; i < 8; ++i) { afrag[i] = alo[i]; afrag[i + 8] = ahi[i]; }
        // B layout: lanes 0-15 hold K=kbase..+15, lanes 16-31 hold K=kbase+16..+31
        v16h bfrag = *(const v16h*)(bcol + kbase + 16 * kh);
        acc = __builtin_amdgcn_wmma_f32_16x16x32_f16(false, afrag, false, bfrag,
                                                     (short)0, acc, false, false);
    }
    // C layout: VGPR r -> row m0 + r + 8*kh, col = lane&15 within tile
    float* crow = C + (size_t)(m0 + 8 * kh) * HID + col;
#pragma unroll
    for (int r = 0; r < 8; ++r) crow[(size_t)r * HID] = acc[r];
}

// ---------------- edge aggregation: one wave per edge, float4 per lane ----------------
__global__ void __launch_bounds__(256)
k_agg(const float* __restrict__ h, const int* __restrict__ ei,
      const float* __restrict__ dis, float* __restrict__ agg, int E) {
    int gw   = (int)((blockIdx.x * blockDim.x + threadIdx.x) >> 5);
    int lane = threadIdx.x & 31;
    if (gw >= E) return;
    int s = ei[gw];
    int d = ei[E + gw];
    float w = dis[s] * dis[d];
    float4 v = ((const float4*)(h + (size_t)s * HID))[lane];
    float* ap = agg + (size_t)d * HID + lane * 4;
    atomicAdd(ap + 0, v.x * w);
    atomicAdd(ap + 1, v.y * w);
    atomicAdd(ap + 2, v.z * w);
    atomicAdd(ap + 3, v.w * w);
}

// ---------------- self-loop + bias + ReLU + f16 convert (fused) ----------------
__global__ void k_finish(float* __restrict__ agg, const float* __restrict__ h,
                         const float* __restrict__ dis, const float* __restrict__ b,
                         _Float16* __restrict__ act16, int n) {
    int idx = blockIdx.x * blockDim.x + threadIdx.x;  // n*HID
    if (idx >= n * HID) return;
    int i = idx >> 7, c = idx & (HID - 1);
    float dii = dis[i];
    float v = agg[idx] + h[idx] * dii * dii + b[c];
    v = fmaxf(v, 0.0f);
    agg[idx]   = v;              // f32 activation (ping-pong buffer; pooled after layer 3)
    act16[idx] = (_Float16)v;    // f16 activation for next WMMA GEMM
}

// ---------------- global mean pool: wave per node ----------------
__global__ void __launch_bounds__(256)
k_pool(const float* __restrict__ act, const int* __restrict__ batch,
       float* __restrict__ psum, float* __restrict__ pcnt, int n) {
    int gw   = (int)((blockIdx.x * blockDim.x + threadIdx.x) >> 5);
    int lane = threadIdx.x & 31;
    if (gw >= n) return;
    int g = batch[gw];
    float4 v = ((const float4*)(act + (size_t)gw * HID))[lane];
    float* p = psum + (size_t)g * HID + lane * 4;
    atomicAdd(p + 0, v.x);
    atomicAdd(p + 1, v.y);
    atomicAdd(p + 2, v.z);
    atomicAdd(p + 3, v.w);
    if (lane == 0) atomicAdd(&pcnt[g], 1.0f);
}

// ---------------- MLP head (64x128 @ 128x64 -> relu -> @ 64x1), single block ----------------
__global__ void __launch_bounds__(256)
k_head(const float* __restrict__ psum, const float* __restrict__ pcnt,
       const float* __restrict__ fc1w, const float* __restrict__ fc1b,
       const float* __restrict__ fc2w, const float* __restrict__ fc2b,
       float* __restrict__ out) {
    __shared__ float pooled[N_GRAPHS * HID];  // 32 KB
    __shared__ float z[N_GRAPHS * 64];        // 16 KB
    int tid = threadIdx.x;
    for (int idx = tid; idx < N_GRAPHS * HID; idx += 256) {
        int g = idx >> 7;
        pooled[idx] = psum[idx] / fmaxf(pcnt[g], 1.0f);
    }
    __syncthreads();
    for (int idx = tid; idx < N_GRAPHS * 64; idx += 256) {
        int g = idx >> 6, j = idx & 63;
        float acc = fc1b[j];
#pragma unroll 8
        for (int k = 0; k < HID; ++k) acc += pooled[g * HID + k] * fc1w[k * 64 + j];
        z[idx] = fmaxf(acc, 0.0f);
    }
    __syncthreads();
    if (tid < N_GRAPHS) {
        float acc = fc2b[0];
#pragma unroll 8
        for (int j = 0; j < 64; ++j) acc += z[tid * 64 + j] * fc2w[j];
        out[tid] = acc;
    }
}

extern "C" void kernel_launch(void* const* d_in, const int* in_sizes, int n_in,
                              void* d_out, int out_size, void* d_ws, size_t ws_size,
                              hipStream_t stream) {
    const float* x     = (const float*)d_in[0];
    const int*   ei    = (const int*)  d_in[1];
    const int*   batch = (const int*)  d_in[2];
    const float* W1    = (const float*)d_in[3];
    const float* b1    = (const float*)d_in[4];
    const float* W2    = (const float*)d_in[5];
    const float* b2    = (const float*)d_in[6];
    const float* W3    = (const float*)d_in[7];
    const float* b3    = (const float*)d_in[8];
    const float* fc1w  = (const float*)d_in[9];
    const float* fc1b  = (const float*)d_in[10];
    const float* fc2w  = (const float*)d_in[11];
    const float* fc2b  = (const float*)d_in[12];
    float* out = (float*)d_out;

    // workspace carve-out (256B aligned)
    char* ws = (char*)d_ws;
    size_t off = 0;
    auto carve = [&](size_t bytes) -> char* {
        char* p = ws + off;
        off += (bytes + 255) & ~(size_t)255;
        return p;
    };
    float*    deg   = (float*)   carve((size_t)N_NODES * 4);
    float*    dis   = (float*)   carve((size_t)N_NODES * 4);
    float*    h     = (float*)   carve((size_t)N_NODES * HID * 4);   // GEMM output
    float*    agg   = (float*)   carve((size_t)N_NODES * HID * 4);   // aggregate / activation
    _Float16* act16 = (_Float16*)carve((size_t)N_NODES * HID * 2);   // f16 activation for WMMA
    _Float16* wt2   = (_Float16*)carve((size_t)HID * HID * 2);
    _Float16* wt3   = (_Float16*)carve((size_t)HID * HID * 2);
    float*    psum  = (float*)   carve((size_t)N_GRAPHS * HID * 4);
    float*    pcnt  = (float*)   carve((size_t)N_GRAPHS * 4);
    (void)ws_size; (void)in_sizes; (void)n_in; (void)out_size;

    const int TB = 256;
    const int gN    = (N_NODES + TB - 1) / TB;
    const int gE    = (N_EDGES + TB - 1) / TB;
    const int gNH   = (N_NODES * HID + TB - 1) / TB;   // 25000
    const int gEW   = (N_EDGES + 7) / 8;               // wave per edge, 8 waves/block
    const int gNW   = (N_NODES + 7) / 8;               // wave per node
    const int gTile = N_NODES / 16;                    // 3125 exact tiles

    // degree / normalization
    k_init_deg<<<gN, TB, 0, stream>>>(deg, N_NODES);
    k_deg<<<gE, TB, 0, stream>>>(ei, deg, N_EDGES);
    k_dis<<<gN, TB, 0, stream>>>(deg, dis, N_NODES);

    // weight prep (f16, transposed)
    k_prep_w<<<(HID * HID + TB - 1) / TB, TB, 0, stream>>>(W2, wt2);
    k_prep_w<<<(HID * HID + TB - 1) / TB, TB, 0, stream>>>(W3, wt3);

    // ---- layer 1 ----
    k_gemm_in<<<gNH, TB, 0, stream>>>(x, W1, h, N_NODES);
    hipMemsetAsync(agg, 0, (size_t)N_NODES * HID * 4, stream);
    k_agg<<<gEW, TB, 0, stream>>>(h, ei, dis, agg, N_EDGES);
    k_finish<<<gNH, TB, 0, stream>>>(agg, h, dis, b1, act16, N_NODES);

    // ---- layer 2 (WMMA) ----
    k_gemm_wmma<<<gTile, TB, 0, stream>>>(act16, wt2, h, N_NODES);
    hipMemsetAsync(agg, 0, (size_t)N_NODES * HID * 4, stream);
    k_agg<<<gEW, TB, 0, stream>>>(h, ei, dis, agg, N_EDGES);
    k_finish<<<gNH, TB, 0, stream>>>(agg, h, dis, b2, act16, N_NODES);

    // ---- layer 3 (WMMA) ----
    k_gemm_wmma<<<gTile, TB, 0, stream>>>(act16, wt3, h, N_NODES);
    hipMemsetAsync(agg, 0, (size_t)N_NODES * HID * 4, stream);
    k_agg<<<gEW, TB, 0, stream>>>(h, ei, dis, agg, N_EDGES);
    k_finish<<<gNH, TB, 0, stream>>>(agg, h, dis, b3, act16, N_NODES);

    // ---- global mean pool + head ----
    hipMemsetAsync(psum, 0, (size_t)N_GRAPHS * HID * 4, stream);
    hipMemsetAsync(pcnt, 0, (size_t)N_GRAPHS * 4, stream);
    k_pool<<<gNW, TB, 0, stream>>>(agg, batch, psum, pcnt, N_NODES);
    k_head<<<1, TB, 0, stream>>>(psum, pcnt, fc1w, fc1b, fc2w, fc2b, out);
}
